// MultiHeadAttention_34368328303207
// MI455X (gfx1250) — matrix operand order repaired and checked by reading, not compiled, not run on hardware
//
#include <hip/hip_runtime.h>
#include <hip/hip_bf16.h>

// ---------------------------------------------------------------------------
// MHA forward for MI455X (gfx1250, wave32): bf16 WMMA 16x16x32 w/ f32 accum,
// TDM (tensor_load_to_lds) double-buffered K/V tile streaming in attention.
// ---------------------------------------------------------------------------

typedef __attribute__((ext_vector_type(16))) __bf16 v16bf;
typedef __attribute__((ext_vector_type(8)))  __bf16 v8bf;
typedef __attribute__((ext_vector_type(4)))  __bf16 v4bf;
typedef __attribute__((ext_vector_type(8)))  float  v8f;
typedef unsigned int u32x4 __attribute__((ext_vector_type(4)));
typedef int          i32x4 __attribute__((ext_vector_type(4)));
typedef int          i32x8 __attribute__((ext_vector_type(8)));

#define WMMA_BF16(a, b, c) \
  __builtin_amdgcn_wmma_f32_16x16x32_bf16(false, (a), false, (b), (short)0, (c), false, false)

#if defined(__has_builtin)
#if __has_builtin(__builtin_amdgcn_tensor_load_to_lds) && \
    __has_builtin(__builtin_amdgcn_s_wait_tensorcnt)
#define HAVE_TDM 1
#endif
#endif

static constexpr int B_  = 4;
static constexpr int S_  = 2048;
static constexpr int D_  = 1024;
static constexpr int H_  = 16;
static constexpr int HD_ = 64;

__device__ __forceinline__ v8f vzero8() {
  v8f z;
#pragma unroll
  for (int i = 0; i < 8; ++i) z[i] = 0.0f;
  return z;
}

// B-operand fragment (K x 16, NT layout): lane (l&15) = column n, 16 contiguous
// K elements starting at k0 (caller adds (lane>>4)*16).
__device__ __forceinline__ v16bf load_bfrag(const __bf16* __restrict__ base,
                                            int nrow, int ld, int k0) {
  const __bf16* p = base + (size_t)nrow * ld + k0;
  v8bf lo = *reinterpret_cast<const v8bf*>(p);
  v8bf hi = *reinterpret_cast<const v8bf*>(p + 8);
  v16bf r;
#pragma unroll
  for (int i = 0; i < 8; ++i) { r[i] = lo[i]; r[i + 8] = hi[i]; }
  return r;
}

// A-operand fragment (16 x 32): lane (l&15) = row, elems 0..15 at
// K = kk+hi*16+0..15 (contiguous, matching load_bfrag).
__device__ __forceinline__ v16bf load_afrag(const __bf16* __restrict__ base,
                                            int row, int ld, int kk, int hi) {
  const __bf16* p = base + (size_t)row * ld + kk + hi * 16;
  v8bf c0 = *reinterpret_cast<const v8bf*>(p);
  v8bf c1 = *reinterpret_cast<const v8bf*>(p + 8);
  v16bf r;
#pragma unroll
  for (int i = 0; i < 8; ++i) { r[i] = c0[i]; r[i + 8] = c1[i]; }
  return r;
}

#ifdef HAVE_TDM
// ---------------------------------------------------------------------------
// TDM: async 2D tile DMA global->LDS. D# packing per CDNA5 ISA ch.8:
//  group0: [1:0]=count=1, [63:32]=lds_addr, [120:64]=global_addr, [127:126]=2
//  group1: [17:16]=data_size(1 => 2B), [79:48]=tensor_dim0, [111:80]=tensor_dim1,
//          [127:112]=tile_dim0, [143:128]=tile_dim1, [207:160]=dim0_stride
// All dims/strides in data_size units. 2D => descriptor groups 2/3 zero.
// ---------------------------------------------------------------------------
__device__ __forceinline__ void tdm_load_2d(const __bf16* gtile, unsigned lds_byte,
                                            unsigned tensor_d0, unsigned tensor_d1,
                                            unsigned tile_d0, unsigned tile_d1,
                                            unsigned long long stride0) {
  unsigned long long ga = (unsigned long long)(uintptr_t)gtile;
  u32x4 g0;
  g0[0] = 1u;                                            // count=1 (valid user D#)
  g0[1] = lds_byte;                                      // lds_addr (bytes)
  g0[2] = (unsigned)(ga & 0xFFFFFFFFu);                  // global_addr[31:0]
  g0[3] = (unsigned)((ga >> 32) & 0x01FFFFFFu) | (2u << 30);  // addr[56:32] | type=2
  i32x8 g1;
  g1[0] = (int)(1u << 16);                               // data_size = 2 bytes
  g1[1] = (int)((tensor_d0 & 0xFFFFu) << 16);            // tensor_dim0[15:0]
  g1[2] = (int)((tensor_d0 >> 16) | ((tensor_d1 & 0xFFFFu) << 16));
  g1[3] = (int)((tensor_d1 >> 16) | (tile_d0 << 16));    // | tile_dim0
  g1[4] = (int)tile_d1;                                  // tile_dim1 (tile_dim2=0)
  g1[5] = (int)(stride0 & 0xFFFFFFFFull);                // dim0_stride[31:0]
  g1[6] = (int)((stride0 >> 32) & 0xFFFFull);            // dim0_stride[47:32]
  g1[7] = 0;
  i32x4 z4 = {0, 0, 0, 0};
#if __clang_major__ >= 23
  i32x8 z8 = {0, 0, 0, 0, 0, 0, 0, 0};
  __builtin_amdgcn_tensor_load_to_lds(g0, g1, z4, z4, z8, 0);
#else
  __builtin_amdgcn_tensor_load_to_lds(g0, g1, z4, z4, 0);
#endif
}
#define WAIT_TENSORCNT(n) do { __builtin_amdgcn_s_wait_tensorcnt(n); \
                               asm volatile("" ::: "memory"); } while (0)
#endif  // HAVE_TDM

// ---------------------------------------------------------------------------
// fp32 -> bf16 conversion (vectorized by 4)
// ---------------------------------------------------------------------------
__global__ void cvt_f32_bf16(const float* __restrict__ src,
                             __bf16* __restrict__ dst, int n4) {
  int i = blockIdx.x * blockDim.x + threadIdx.x;
  int stride = gridDim.x * blockDim.x;
  for (; i < n4; i += stride) {
    float4 f = reinterpret_cast<const float4*>(src)[i];
    v4bf o;
    o[0] = (__bf16)f.x; o[1] = (__bf16)f.y;
    o[2] = (__bf16)f.z; o[3] = (__bf16)f.w;
    reinterpret_cast<v4bf*>(dst)[i] = o;
  }
}

// ---------------------------------------------------------------------------
// Fused QKV projection: per wave a 16x64 tile of Q, K, V (12 accumulators).
// Q pre-scaled by 1/sqrt(D). K stored [B,H,S,HD]; V stored transposed
// [B,H,HD,S] so attention's PV matmul is NT-contiguous.
// ---------------------------------------------------------------------------
__global__ void qkv_proj_kernel(const __bf16* __restrict__ xb,
                                const __bf16* __restrict__ wq,
                                const __bf16* __restrict__ wk,
                                const __bf16* __restrict__ wv,
                                const float* __restrict__ bq,
                                const float* __restrict__ bk,
                                const float* __restrict__ bv,
                                __bf16* __restrict__ Qo,
                                __bf16* __restrict__ Ko,
                                __bf16* __restrict__ Vto) {
  const int lane = threadIdx.x & 31;
  const int wave = threadIdx.x >> 5;
  const int ln = lane & 15, hi = lane >> 4;
  const int gw = blockIdx.x * 4 + wave;          // 8192 tiles total
  const int mBase = (gw >> 4) * 16;              // 512 m-tiles
  const int nBase = (gw & 15) * 64;              // 16 n-tiles of 64

  v8f accQ[4], accK[4], accV[4];
#pragma unroll
  for (int b = 0; b < 4; ++b) { accQ[b] = vzero8(); accK[b] = vzero8(); accV[b] = vzero8(); }

  const __bf16* arow = xb + (size_t)(mBase + ln) * D_;
  for (int kk = 0; kk < D_; kk += 32) {
    if (kk + 32 < D_) __builtin_prefetch(arow + kk + 32, 0, 3);
    v16bf aF = load_afrag(xb, mBase + ln, D_, kk, hi);
    const int k0 = kk + hi * 16;
#pragma unroll
    for (int b = 0; b < 4; ++b) {
      const int nrow = nBase + b * 16 + ln;
      // issue all three B-fragment load clauses before the WMMAs
      v16bf fq = load_bfrag(wq, nrow, D_, k0);
      v16bf fk = load_bfrag(wk, nrow, D_, k0);
      v16bf fv = load_bfrag(wv, nrow, D_, k0);
      accQ[b] = WMMA_BF16(aF, fq, accQ[b]);
      accK[b] = WMMA_BF16(aF, fk, accK[b]);
      accV[b] = WMMA_BF16(aF, fv, accV[b]);
    }
  }

  const float qscale = 0.03125f;  // 1/sqrt(1024), folded into Q
#pragma unroll
  for (int b = 0; b < 4; ++b) {
    const int col = nBase + b * 16 + ln;
    const float vbq = bq[col], vbk = bk[col], vbv = bv[col];
    const int h = col >> 6, hd = col & 63;
#pragma unroll
    for (int g = 0; g < 8; ++g) {
      const int row = mBase + g + 8 * hi;      // global (b,s) row
      const int bt = row >> 11, s = row & (S_ - 1);
      const size_t qk = (((size_t)(bt * H_ + h)) * S_ + s) * HD_ + hd;
      Qo[qk] = (__bf16)((accQ[b][g] + vbq) * qscale);
      Ko[qk] = (__bf16)(accK[b][g] + vbk);
      Vto[(((size_t)(bt * H_ + h)) * HD_ + hd) * S_ + s] = (__bf16)(accV[b][g] + vbv);
    }
  }
}

// ---------------------------------------------------------------------------
// Flash attention: one wave per 16-query tile, k-blocks of 32, online softmax.
// TDM path: K (32x64) and V (64x32) tiles DMA'd to LDS, double-buffered with
// counted s_wait_tensorcnt; WMMA operands come from LDS. P staged through LDS
// to convert WMMA C-layout -> A-fragment layout.
// ---------------------------------------------------------------------------
__global__ void attn_kernel(const __bf16* __restrict__ Q,
                            const __bf16* __restrict__ K,
                            const __bf16* __restrict__ Vt,
                            __bf16* __restrict__ attnB) {
#ifdef HAVE_TDM
  __shared__ __align__(16) __bf16 kbuf[4][2][32 * 64];
  __shared__ __align__(16) __bf16 vbuf[4][2][64 * 32];
#endif
  __shared__ __align__(16) __bf16 pbuf[4][16 * 32];

  const int lane = threadIdx.x & 31;
  const int wave = threadIdx.x >> 5;
  const int ln = lane & 15, hi = lane >> 4;
  const int gw = blockIdx.x * 4 + wave;        // 8192 tiles = B*H*(S/16)
  const int bh = gw >> 7;                      // 64 (b,h) pairs
  const int qBase = (gw & 127) * 16;

  const __bf16* Qh = Q  + (size_t)bh * S_ * HD_;
  const __bf16* Kh = K  + (size_t)bh * S_ * HD_;
  const __bf16* Vh = Vt + (size_t)bh * HD_ * S_;

  // Q A-fragments for both 32-wide d-chunks, loaded once.
  const v16bf aQ0 = load_afrag(Qh, qBase + ln, HD_, 0, hi);
  const v16bf aQ1 = load_afrag(Qh, qBase + ln, HD_, 32, hi);

  v8f Oacc[4];
#pragma unroll
  for (int b = 0; b < 4; ++b) Oacc[b] = vzero8();
  v8f mrow, lrow;
#pragma unroll
  for (int g = 0; g < 8; ++g) { mrow[g] = -1.0e30f; lrow[g] = 0.0f; }

  __bf16* sp = pbuf[wave];
  const int nsteps = (qBase + 16 + 31) / 32;

#ifdef HAVE_TDM
  // Prime the pipeline: DMA K/V tiles for step 0.
  tdm_load_2d(Kh, (unsigned)(uintptr_t)kbuf[wave][0],
              HD_, S_, HD_, 32, HD_);                       // K: [32 keys][64 d]
  tdm_load_2d(Vh, (unsigned)(uintptr_t)vbuf[wave][0],
              S_, HD_, 32, HD_, S_);                        // V: [64 d][32 keys]
#endif

  for (int it = 0; it < nsteps; ++it) {
    const int kb = it * 32;

#ifdef HAVE_TDM
    if (it + 1 < nsteps) {
      const int nb = (it + 1) & 1;
      tdm_load_2d(Kh + (size_t)(kb + 32) * HD_, (unsigned)(uintptr_t)kbuf[wave][nb],
                  HD_, S_, HD_, 32, HD_);
      tdm_load_2d(Vh + (kb + 32), (unsigned)(uintptr_t)vbuf[wave][nb],
                  S_, HD_, 32, HD_, S_);
      WAIT_TENSORCNT(2);   // current tiles done; next two DMAs still in flight
    } else {
      WAIT_TENSORCNT(0);
    }
    const __bf16* kt = kbuf[wave][it & 1];   // [32][64], local key rows 0..31
    const __bf16* vt = vbuf[wave][it & 1];   // [64][32], local k cols 0..31
#else
    const __bf16* kt = Kh + (size_t)kb * HD_;   // fallback: straight from global
    const __bf16* vt = Vh;                       // (column offset handled below)
#endif

    // scores: two 16x16 C-tiles covering keys [kb, kb+32)
    v8f s0 = vzero8(), s1 = vzero8();
    {
      v16bf b00 = load_bfrag(kt, ln,      HD_, hi * 16);
      v16bf b01 = load_bfrag(kt, ln,      HD_, 32 + hi * 16);
      v16bf b10 = load_bfrag(kt, 16 + ln, HD_, hi * 16);
      v16bf b11 = load_bfrag(kt, 16 + ln, HD_, 32 + hi * 16);
      s0 = WMMA_BF16(aQ0, b00, s0);
      s0 = WMMA_BF16(aQ1, b01, s0);
      s1 = WMMA_BF16(aQ0, b10, s1);
      s1 = WMMA_BF16(aQ1, b11, s1);
    }

    if (kb + 31 > qBase) {  // causal mask only on boundary blocks
#pragma unroll
      for (int g = 0; g < 8; ++g) {
        const int qrow = qBase + g + 8 * hi;
        if (kb + ln      > qrow) s0[g] = -1.0e30f;
        if (kb + 16 + ln > qrow) s1[g] = -1.0e30f;
      }
    }

    // online softmax (row stats: reduce across the 16 lanes of each half)
    v8f mnew, p0, p1, fsc, rs;
#pragma unroll
    for (int g = 0; g < 8; ++g) mnew[g] = fmaxf(s0[g], s1[g]);
#pragma unroll
    for (int mk = 1; mk <= 8; mk <<= 1)
#pragma unroll
      for (int g = 0; g < 8; ++g) mnew[g] = fmaxf(mnew[g], __shfl_xor(mnew[g], mk, 32));
#pragma unroll
    for (int g = 0; g < 8; ++g) {
      mnew[g] = fmaxf(mnew[g], mrow[g]);
      fsc[g] = __expf(mrow[g] - mnew[g]);
      p0[g]  = __expf(s0[g] - mnew[g]);
      p1[g]  = __expf(s1[g] - mnew[g]);
      rs[g]  = p0[g] + p1[g];
      mrow[g] = mnew[g];
    }
#pragma unroll
    for (int mk = 1; mk <= 8; mk <<= 1)
#pragma unroll
      for (int g = 0; g < 8; ++g) rs[g] += __shfl_xor(rs[g], mk, 32);
#pragma unroll
    for (int g = 0; g < 8; ++g) lrow[g] = lrow[g] * fsc[g] + rs[g];
#pragma unroll
    for (int b = 0; b < 4; ++b)
#pragma unroll
      for (int g = 0; g < 8; ++g) Oacc[b][g] *= fsc[g];

    // stage P (C-layout) -> LDS row-major 16x32 -> reload as A-fragment
    asm volatile("" ::: "memory");
#pragma unroll
    for (int g = 0; g < 8; ++g) {
      const int Mr = g + 8 * hi;
      sp[Mr * 32 + ln]      = (__bf16)p0[g];
      sp[Mr * 32 + 16 + ln] = (__bf16)p1[g];
    }
    asm volatile("" ::: "memory");
    const v16bf aP = load_afrag(sp, ln, 32, 0, hi);
    asm volatile("" ::: "memory");

    // O += P @ V
#ifdef HAVE_TDM
#pragma unroll
    for (int b = 0; b < 4; ++b)
      Oacc[b] = WMMA_BF16(aP, load_bfrag(vt, b * 16 + ln, 32, hi * 16), Oacc[b]);
#else
#pragma unroll
    for (int b = 0; b < 4; ++b)
      Oacc[b] = WMMA_BF16(aP, load_bfrag(vt, b * 16 + ln, S_, kb + hi * 16), Oacc[b]);
#endif
  }

  // normalize and write merged-head bf16 activations [B,S,D]
  const int bt = bh >> 4, h = bh & 15;
#pragma unroll
  for (int b = 0; b < 4; ++b) {
    const int d = b * 16 + ln;
#pragma unroll
    for (int g = 0; g < 8; ++g) {
      const int s = qBase + g + 8 * hi;
      const float o = Oacc[b][g] / lrow[g];
      attnB[((size_t)bt * S_ + s) * D_ + h * HD_ + d] = (__bf16)o;
    }
  }
}

// ---------------------------------------------------------------------------
// Output projection: out = attn @ Wo^T + bo  (fp32 output)
// ---------------------------------------------------------------------------
__global__ void out_proj_kernel(const __bf16* __restrict__ ab,
                                const __bf16* __restrict__ wo,
                                const float* __restrict__ bo,
                                float* __restrict__ out) {
  const int lane = threadIdx.x & 31;
  const int wave = threadIdx.x >> 5;
  const int ln = lane & 15, hi = lane >> 4;
  const int gw = blockIdx.x * 4 + wave;
  const int mBase = (gw >> 4) * 16;
  const int nBase = (gw & 15) * 64;

  v8f acc[4];
#pragma unroll
  for (int b = 0; b < 4; ++b) acc[b] = vzero8();

  const __bf16* arow = ab + (size_t)(mBase + ln) * D_;
  for (int kk = 0; kk < D_; kk += 32) {
    if (kk + 32 < D_) __builtin_prefetch(arow + kk + 32, 0, 3);
    v16bf aF = load_afrag(ab, mBase + ln, D_, kk, hi);
    const int k0 = kk + hi * 16;
    v16bf f0 = load_bfrag(wo, nBase + 0 * 16 + ln, D_, k0);
    v16bf f1 = load_bfrag(wo, nBase + 1 * 16 + ln, D_, k0);
    v16bf f2 = load_bfrag(wo, nBase + 2 * 16 + ln, D_, k0);
    v16bf f3 = load_bfrag(wo, nBase + 3 * 16 + ln, D_, k0);
    acc[0] = WMMA_BF16(aF, f0, acc[0]);
    acc[1] = WMMA_BF16(aF, f1, acc[1]);
    acc[2] = WMMA_BF16(aF, f2, acc[2]);
    acc[3] = WMMA_BF16(aF, f3, acc[3]);
  }

#pragma unroll
  for (int b = 0; b < 4; ++b) {
    const int col = nBase + b * 16 + ln;
    const float vb = bo[col];
#pragma unroll
    for (int g = 0; g < 8; ++g) {
      const int row = mBase + g + 8 * hi;
      out[(size_t)row * D_ + col] = acc[b][g] + vb;
    }
  }
}

// ---------------------------------------------------------------------------
extern "C" void kernel_launch(void* const* d_in, const int* in_sizes, int n_in,
                              void* d_out, int out_size, void* d_ws, size_t ws_size,
                              hipStream_t stream) {
  (void)in_sizes; (void)n_in; (void)out_size; (void)ws_size;
  const float* x  = (const float*)d_in[0];
  const float* Wq = (const float*)d_in[1];
  const float* bq = (const float*)d_in[2];
  const float* Wk = (const float*)d_in[3];
  const float* bk = (const float*)d_in[4];
  const float* Wv = (const float*)d_in[5];
  const float* bv = (const float*)d_in[6];
  const float* Wo = (const float*)d_in[7];
  const float* bo = (const float*)d_in[8];
  float* out = (float*)d_out;

  char* ws = (char*)d_ws;
  const size_t MB = 1024 * 1024;
  __bf16* xb  = (__bf16*)(ws +  0 * MB);   // 16 MB
  __bf16* wqb = (__bf16*)(ws + 16 * MB);   //  2 MB
  __bf16* wkb = (__bf16*)(ws + 18 * MB);   //  2 MB
  __bf16* wvb = (__bf16*)(ws + 20 * MB);   //  2 MB
  __bf16* wob = (__bf16*)(ws + 22 * MB);   //  2 MB
  __bf16* Qb  = (__bf16*)(ws + 24 * MB);   // 16 MB  [B,H,S,HD], pre-scaled
  __bf16* Kb  = (__bf16*)(ws + 40 * MB);   // 16 MB  [B,H,S,HD]
  __bf16* Vtb = (__bf16*)(ws + 56 * MB);   // 16 MB  [B,H,HD,S]
  __bf16* abf = (__bf16*)(ws + 72 * MB);   // 16 MB  merged heads [B,S,D]

  cvt_f32_bf16<<<4096, 256, 0, stream>>>(x,  xb,  (B_ * S_ * D_) / 4);
  cvt_f32_bf16<<<1024, 256, 0, stream>>>(Wq, wqb, (D_ * D_) / 4);
  cvt_f32_bf16<<<1024, 256, 0, stream>>>(Wk, wkb, (D_ * D_) / 4);
  cvt_f32_bf16<<<1024, 256, 0, stream>>>(Wv, wvb, (D_ * D_) / 4);
  cvt_f32_bf16<<<1024, 256, 0, stream>>>(Wo, wob, (D_ * D_) / 4);

  // 8192 wave-tiles / 4 waves per block
  qkv_proj_kernel<<<2048, 128, 0, stream>>>(xb, wqb, wkb, wvb, bq, bk, bv, Qb, Kb, Vtb);
  attn_kernel<<<2048, 128, 0, stream>>>(Qb, Kb, Vtb, abf);
  out_proj_kernel<<<2048, 128, 0, stream>>>(abf, wob, bo, out);
}
